// TimeVAEDecoder_9148280340656
// MI455X (gfx1250) — matrix operand order repaired and checked by reading, not compiled, not run on hardware
//
#include <hip/hip_runtime.h>
#include <hip/hip_bf16.h>
#include <stdint.h>

#define B_   128
#define LAT_ 128
#define H_   512
#define F_   128
#define L_   256
#define NL_  2
#define G4_  2048   // 4*H

typedef __bf16 bf16;
typedef __attribute__((ext_vector_type(16))) bf16  v16bf;
typedef __attribute__((ext_vector_type(8)))  float v8f;

#if defined(__has_builtin)
#if __has_builtin(__builtin_amdgcn_sched_barrier)
#define SCHED_FENCE() __builtin_amdgcn_sched_barrier(0)
#endif
#endif
#ifndef SCHED_FENCE
#define SCHED_FENCE()
#endif

// ---------- scalar helpers ----------
static __device__ __forceinline__ bf16 f2bf(float f) {
  union { float f; uint32_t u; } x; x.f = f;
  uint32_t r = (x.u + 0x7FFFu + ((x.u >> 16) & 1u)) >> 16;   // round-nearest-even
  union { uint16_t s; bf16 b; } y; y.s = (uint16_t)r;
  return y.b;
}
static __device__ __forceinline__ float bf2f(bf16 b) {
  union { uint16_t s; bf16 b; } y; y.b = b;
  union { float f; uint32_t u; } x; x.u = ((uint32_t)y.s) << 16;
  return x.f;
}
static __device__ __forceinline__ float sigm(float x) { return 1.f / (1.f + __expf(-x)); }

// ---------- WMMA fragment loaders (CDNA5 16-bit layouts, wave32) ----------
// A 16x32 (MxK): lanes 0-15 -> M rows, half=lane>>4 selects K sub-block.
static __device__ __forceinline__ v16bf load_a_lds(const bf16* base, int strideElems,
                                                   int m, int kbase, int lhalf) {
  union { v16bf v; uint4 u[2]; } r;
  const bf16* p = base + m * strideElems + kbase + lhalf * 8;
  r.u[0] = *(const uint4*)(p);
  r.u[1] = *(const uint4*)(p + 16);
  return r.v;
}
// B 32x16 (KxN): lane&15 -> N column = weight row (W row-major [out][in]); K = 8*half+[0,8) and 16+8*half+[0,8).
static __device__ __forceinline__ v16bf load_b_glb(const bf16* W, int Kw,
                                                   int nrow, int kbase, int lhalf) {
  union { v16bf v; uint4 u[2]; } r;
  const bf16* p = W + (size_t)nrow * Kw + kbase + lhalf * 8;
  r.u[0] = *(const uint4*)(p);
  r.u[1] = *(const uint4*)(p + 16);
  return r.v;
}
static __device__ __forceinline__ v8f wmma_bf16(v16bf a, v16bf b, v8f c) {
  return __builtin_amdgcn_wmma_f32_16x16x32_bf16(false, a, false, b, (short)0, c, false, false);
}
// C-init: 8 consecutive f32 per lane (rows m=0..7 / 8..15 of transposed bias table)
static __device__ __forceinline__ v8f load_c8(const float* p) {
  union { v8f v; float4 u[2]; } r;
  r.u[0] = *(const float4*)(p);
  r.u[1] = *(const float4*)(p + 4);
  return r.v;
}

// ---------- prologue kernels ----------
__global__ void cvt_bf16_kernel(const float* __restrict__ src, bf16* __restrict__ dst, int n) {
  int i = blockIdx.x * blockDim.x + threadIdx.x;
  if (i < n) dst[i] = f2bf(src[i]);
}

// Torch-faithful (B, NL*H) -> reshape(NL, B, H):  h0[l][b][h] = M[l*64 + (b>>1)][(b&1)*512 + h]
__global__ void init_hc_kernel(const float* __restrict__ z,
                               const float* __restrict__ Wl2h, const float* __restrict__ bl2h,
                               const float* __restrict__ Wl2c, const float* __restrict__ bl2c,
                               float* __restrict__ haF, float* __restrict__ caF,
                               float* __restrict__ hbF, float* __restrict__ cbF) {
  int i = blockIdx.x * blockDim.x + threadIdx.x;
  if (i >= NL_ * B_ * H_) return;
  int l = i >> 16; int r = i & 65535; int b = r >> 9; int h = r & 511;
  int zr  = l * 64 + (b >> 1);
  int col = ((b & 1) << 9) + h;
  float s1 = bl2h[col], s2 = bl2c[col];
  for (int k = 0; k < LAT_; ++k) {
    float zv = z[zr * LAT_ + k];
    s1 += zv * Wl2h[col * LAT_ + k];
    s2 += zv * Wl2c[col * LAT_ + k];
  }
  if (l == 0) { haF[b * H_ + h] = s1; caF[b * H_ + h] = s2; }
  else        { hbF[b * H_ + h] = s1; cbF[b * H_ + h] = s2; }
}

__global__ void zproj_kernel(const float* __restrict__ z, const float* __restrict__ Wzp,
                             const float* __restrict__ bzp, float* __restrict__ zpF) {
  int i = blockIdx.x * blockDim.x + threadIdx.x;
  if (i >= B_ * H_) return;
  int b = i >> 9, h = i & 511;
  float s = bzp[h];
  for (int k = 0; k < LAT_; ++k) s += z[b * LAT_ + k] * Wzp[h * LAT_ + k];
  zpF[i] = s;
}

// gb0T[n][b] = z_proj[b] . W_ih0[n][128:640] + b_ih0[n] + b_hh0[n]  (TRANSPOSED: b contiguous)
__global__ void gbias0_kernel(const float* __restrict__ zpF, const float* __restrict__ Wih0,
                              const float* __restrict__ bih0, const float* __restrict__ bhh0,
                              float* __restrict__ gb0T) {
  int i = blockIdx.x * blockDim.x + threadIdx.x;
  if (i >= B_ * G4_) return;
  int b = i >> 11, n = i & 2047;
  float s = bih0[n] + bhh0[n];
  const float* wrow = Wih0 + (size_t)n * 640 + 128;
  for (int k = 0; k < H_; ++k) s += zpF[b * H_ + k] * wrow[k];
  gb0T[(size_t)n * B_ + b] = s;
}

__global__ void cbias1_kernel(const float* __restrict__ bih1, const float* __restrict__ bhh1,
                              float* __restrict__ cb1) {
  int i = blockIdx.x * blockDim.x + threadIdx.x;
  if (i < G4_) cb1[i] = bih1[i] + bhh1[i];
}

// ---------- persistent decoder: 8 blocks x 512 threads (16 waves), M-tile = 16 batch rows ----------
__global__ __launch_bounds__(512, 1)
void decoder_kernel(const bf16* __restrict__ Wih0, const bf16* __restrict__ Whh0,
                    const bf16* __restrict__ Wih1, const bf16* __restrict__ Whh1,
                    const bf16* __restrict__ Wpre, const bf16* __restrict__ Wout,
                    const float* __restrict__ gb0T, const float* __restrict__ cb1,
                    const float* __restrict__ bpre, const float* __restrict__ bout,
                    const float* __restrict__ lng,  const float* __restrict__ lnb,
                    const float* __restrict__ haF,  const float* __restrict__ caF,
                    const float* __restrict__ hbF,  const float* __restrict__ cbF,
                    float* __restrict__ out) {
  __shared__ __align__(16) bf16 A0[16 * 128];    // dec (cell0 A, K=128)
  __shared__ __align__(16) bf16 A1[16 * 1024];   // [ha | hb]
  __shared__ __align__(16) bf16 S[16 * 512];     // ln, then (relu(pre)+ln) in place
  __shared__ float red1[16 * 32];
  __shared__ float red2[16 * 32];
  __shared__ float stats[16 * 2];

  const int tid   = threadIdx.x;
  const int wave  = tid >> 5;
  const int lane  = tid & 31;
  const int lhalf = lane >> 4;
  const int lmod  = lane & 15;
  const int bb    = blockIdx.x * 16;
  const int nw    = wave * 32 + lmod;   // per-wave N base (+ q*512 + t*16) for gate GEMMs
  const int mrow  = 8 * lhalf;          // C-tile row base for this lane

  // cell states in registers, C-tile layout: element (m = j + mrow, col = wave*32 + t*16 + lmod)
  float ca[2][8], cb[2][8];
  #pragma unroll
  for (int t = 0; t < 2; ++t) {
    #pragma unroll
    for (int j = 0; j < 8; ++j) {
      int m = j + mrow, col = nw + t * 16;
      ca[t][j] = caF[(bb + m) * H_ + col];
      cb[t][j] = cbF[(bb + m) * H_ + col];
    }
  }
  for (int i = tid; i < 16 * H_; i += 512) {
    int m = i >> 9, c = i & 511;
    A1[m * 1024 + c]       = f2bf(haF[(bb + m) * H_ + c]);
    A1[m * 1024 + 512 + c] = f2bf(hbF[(bb + m) * H_ + c]);
  }
  for (int i = tid; i < 16 * F_; i += 512) A0[i] = f2bf(0.f);
  __syncthreads();

  for (int step = 0; step < L_; ++step) {
    // ================= cell 0: gates = dec@Wih0[:, :128]^T + ha@Whh0^T + gb0 =================
    v8f acc[8];
    #pragma unroll
    for (int q = 0; q < 4; ++q)
      #pragma unroll
      for (int t = 0; t < 2; ++t)
        acc[q * 2 + t] = load_c8(gb0T + (size_t)(q * 512 + nw + t * 16) * B_ + bb + mrow);
    #pragma unroll 1
    for (int ks = 0; ks < 4; ++ks) {                 // dec part, K=128
      v16bf a = load_a_lds(A0, 128, lmod, ks * 32, lhalf);
      #pragma unroll
      for (int q = 0; q < 4; ++q)
        #pragma unroll
        for (int t = 0; t < 2; ++t) {
          v16bf b = load_b_glb(Wih0, 640, q * 512 + nw + t * 16, ks * 32, lhalf);
          acc[q * 2 + t] = wmma_bf16(a, b, acc[q * 2 + t]);
        }
      SCHED_FENCE();
    }
    #pragma unroll 1
    for (int ks = 0; ks < 16; ++ks) {                // ha part, K=512
      v16bf a = load_a_lds(A1, 1024, lmod, ks * 32, lhalf);
      #pragma unroll
      for (int q = 0; q < 4; ++q)
        #pragma unroll
        for (int t = 0; t < 2; ++t) {
          v16bf b = load_b_glb(Whh0, 512, q * 512 + nw + t * 16, ks * 32, lhalf);
          acc[q * 2 + t] = wmma_bf16(a, b, acc[q * 2 + t]);
        }
      SCHED_FENCE();
    }
    float hn[2][8];
    #pragma unroll
    for (int t = 0; t < 2; ++t)
      #pragma unroll
      for (int j = 0; j < 8; ++j) {
        float iv = sigm(acc[0 + t][j]);
        float fv = sigm(acc[2 + t][j]);
        float gv = tanhf(acc[4 + t][j]);
        float ov = sigm(acc[6 + t][j]);
        float c  = fv * ca[t][j] + iv * gv;
        ca[t][j] = c;
        hn[t][j] = ov * tanhf(c);
      }
    __syncthreads();                                 // everyone done reading old ha
    #pragma unroll
    for (int t = 0; t < 2; ++t) {
      int colb = nw + t * 16;
      #pragma unroll
      for (int j = 0; j < 8; ++j) A1[(j + mrow) * 1024 + colb] = f2bf(hn[t][j]);
    }
    __syncthreads();

    // ================= cell 1: gates = ha@Wih1^T + hb@Whh1^T + cb1 =================
    #pragma unroll
    for (int q = 0; q < 4; ++q)
      #pragma unroll
      for (int t = 0; t < 2; ++t) {
        float bv = cb1[q * 512 + nw + t * 16];
        #pragma unroll
        for (int j = 0; j < 8; ++j) acc[q * 2 + t][j] = bv;
      }
    #pragma unroll 1
    for (int ks = 0; ks < 16; ++ks) {                // ha part
      v16bf a = load_a_lds(A1, 1024, lmod, ks * 32, lhalf);
      #pragma unroll
      for (int q = 0; q < 4; ++q)
        #pragma unroll
        for (int t = 0; t < 2; ++t) {
          v16bf b = load_b_glb(Wih1, 512, q * 512 + nw + t * 16, ks * 32, lhalf);
          acc[q * 2 + t] = wmma_bf16(a, b, acc[q * 2 + t]);
        }
      SCHED_FENCE();
    }
    #pragma unroll 1
    for (int ks = 0; ks < 16; ++ks) {                // hb part
      v16bf a = load_a_lds(A1 + 512, 1024, lmod, ks * 32, lhalf);
      #pragma unroll
      for (int q = 0; q < 4; ++q)
        #pragma unroll
        for (int t = 0; t < 2; ++t) {
          v16bf b = load_b_glb(Whh1, 512, q * 512 + nw + t * 16, ks * 32, lhalf);
          acc[q * 2 + t] = wmma_bf16(a, b, acc[q * 2 + t]);
        }
      SCHED_FENCE();
    }
    #pragma unroll
    for (int t = 0; t < 2; ++t)
      #pragma unroll
      for (int j = 0; j < 8; ++j) {
        float iv = sigm(acc[0 + t][j]);
        float fv = sigm(acc[2 + t][j]);
        float gv = tanhf(acc[4 + t][j]);
        float ov = sigm(acc[6 + t][j]);
        float c  = fv * cb[t][j] + iv * gv;
        cb[t][j] = c;
        hn[t][j] = ov * tanhf(c);
      }
    __syncthreads();                                 // everyone done reading old hb
    #pragma unroll
    for (int t = 0; t < 2; ++t) {
      int colb = 512 + nw + t * 16;
      #pragma unroll
      for (int j = 0; j < 8; ++j) A1[(j + mrow) * 1024 + colb] = f2bf(hn[t][j]);
    }
    __syncthreads();

    // ================= LayerNorm(hb) -> S =================
    {
      int row = tid >> 5, seg = tid & 31;
      float sm = 0.f, sq = 0.f;
      #pragma unroll
      for (int e = 0; e < 16; ++e) {
        float v = bf2f(A1[row * 1024 + 512 + seg * 16 + e]);
        sm += v; sq += v * v;
      }
      red1[row * 32 + seg] = sm;
      red2[row * 32 + seg] = sq;
      __syncthreads();
      if (seg == 0) {
        float ts = 0.f, tq = 0.f;
        for (int k2 = 0; k2 < 32; ++k2) { ts += red1[row * 32 + k2]; tq += red2[row * 32 + k2]; }
        float mean = ts * (1.f / 512.f);
        float var  = tq * (1.f / 512.f) - mean * mean;
        stats[row * 2]     = mean;
        stats[row * 2 + 1] = rsqrtf(var + 1e-5f);
      }
      __syncthreads();
      float mean = stats[row * 2], rs = stats[row * 2 + 1];
      #pragma unroll
      for (int e = 0; e < 16; ++e) {
        int col = seg * 16 + e;
        float v = bf2f(A1[row * 1024 + 512 + col]);
        S[row * 512 + col] = f2bf((v - mean) * rs * lng[col] + lnb[col]);
      }
      __syncthreads();
    }

    // ================= pre = relu(ln @ Wpre^T + bpre); S <- pre + ln (in place) =================
    v8f pacc[2];
    #pragma unroll
    for (int t = 0; t < 2; ++t) {
      float bv = bpre[nw + t * 16];
      #pragma unroll
      for (int j = 0; j < 8; ++j) pacc[t][j] = bv;
    }
    #pragma unroll 1
    for (int ks = 0; ks < 16; ++ks) {
      v16bf a = load_a_lds(S, 512, lmod, ks * 32, lhalf);
      #pragma unroll
      for (int t = 0; t < 2; ++t) {
        v16bf b = load_b_glb(Wpre, 512, nw + t * 16, ks * 32, lhalf);
        pacc[t] = wmma_bf16(a, b, pacc[t]);
      }
      SCHED_FENCE();
    }
    __syncthreads();                                 // all reads of ln done
    #pragma unroll
    for (int t = 0; t < 2; ++t) {
      int n = nw + t * 16;
      #pragma unroll
      for (int j = 0; j < 8; ++j) {
        int m = j + mrow;
        float lnv = bf2f(S[m * 512 + n]);
        S[m * 512 + n] = f2bf(fmaxf(pacc[t][j], 0.f) + lnv);
      }
    }
    __syncthreads();

    // ================= out_t = S @ Wout^T + bout; write global + feed back as dec =================
    if (wave < 8) {
      v8f oacc;
      float bv = bout[wave * 16 + lmod];
      #pragma unroll
      for (int j = 0; j < 8; ++j) oacc[j] = bv;
      #pragma unroll 1
      for (int ks = 0; ks < 16; ++ks) {
        v16bf a = load_a_lds(S, 512, lmod, ks * 32, lhalf);
        v16bf b = load_b_glb(Wout, 512, wave * 16 + lmod, ks * 32, lhalf);
        oacc = wmma_bf16(a, b, oacc);
        SCHED_FENCE();
      }
      #pragma unroll
      for (int j = 0; j < 8; ++j) {
        int m = j + mrow, n = wave * 16 + lmod;
        out[(size_t)(bb + m) * L_ * F_ + (size_t)step * F_ + n] = oacc[j];
        A0[m * 128 + n] = f2bf(oacc[j]);
      }
    }
    __syncthreads();
  }
}

// ---------- launcher ----------
extern "C" void kernel_launch(void* const* d_in, const int* in_sizes, int n_in,
                              void* d_out, int out_size, void* d_ws, size_t ws_size,
                              hipStream_t stream) {
  (void)in_sizes; (void)n_in; (void)out_size; (void)ws_size;
  const float* z    = (const float*)d_in[0];
  const float* Wl2h = (const float*)d_in[1];
  const float* bl2h = (const float*)d_in[2];
  const float* Wl2c = (const float*)d_in[3];
  const float* bl2c = (const float*)d_in[4];
  const float* Wzp  = (const float*)d_in[5];
  const float* bzp  = (const float*)d_in[6];
  const float* Wih0 = (const float*)d_in[7];
  const float* Whh0 = (const float*)d_in[8];
  const float* bih0 = (const float*)d_in[9];
  const float* bhh0 = (const float*)d_in[10];
  const float* Wih1 = (const float*)d_in[11];
  const float* Whh1 = (const float*)d_in[12];
  const float* bih1 = (const float*)d_in[13];
  const float* bhh1 = (const float*)d_in[14];
  const float* lng  = (const float*)d_in[15];
  const float* lnb  = (const float*)d_in[16];
  const float* Wpre = (const float*)d_in[17];
  const float* bpre = (const float*)d_in[18];
  const float* Wout = (const float*)d_in[19];
  const float* bout = (const float*)d_in[20];

  char* ws = (char*)d_ws;
  size_t off = 0;
  auto carve = [&](size_t bytes) -> void* {
    off = (off + 255) & ~(size_t)255;
    void* p = ws + off;
    off += bytes;
    return p;
  };
  bf16* Wih0h = (bf16*)carve((size_t)G4_ * 640 * 2);
  bf16* Whh0h = (bf16*)carve((size_t)G4_ * 512 * 2);
  bf16* Wih1h = (bf16*)carve((size_t)G4_ * 512 * 2);
  bf16* Whh1h = (bf16*)carve((size_t)G4_ * 512 * 2);
  bf16* Wpreh = (bf16*)carve((size_t)512 * 512 * 2);
  bf16* Wouth = (bf16*)carve((size_t)128 * 512 * 2);
  float* haF  = (float*)carve((size_t)B_ * H_ * 4);
  float* caF  = (float*)carve((size_t)B_ * H_ * 4);
  float* hbF  = (float*)carve((size_t)B_ * H_ * 4);
  float* cbF  = (float*)carve((size_t)B_ * H_ * 4);
  float* zpF  = (float*)carve((size_t)B_ * H_ * 4);
  float* gb0T = (float*)carve((size_t)B_ * G4_ * 4);
  float* cb1  = (float*)carve((size_t)G4_ * 4);

  auto cvt = [&](const float* s, bf16* d, int n) {
    cvt_bf16_kernel<<<(n + 255) / 256, 256, 0, stream>>>(s, d, n);
  };
  cvt(Wih0, Wih0h, G4_ * 640);
  cvt(Whh0, Whh0h, G4_ * 512);
  cvt(Wih1, Wih1h, G4_ * 512);
  cvt(Whh1, Whh1h, G4_ * 512);
  cvt(Wpre, Wpreh, 512 * 512);
  cvt(Wout, Wouth, 128 * 512);

  init_hc_kernel<<<(NL_ * B_ * H_ + 255) / 256, 256, 0, stream>>>(
      z, Wl2h, bl2h, Wl2c, bl2c, haF, caF, hbF, cbF);
  zproj_kernel<<<(B_ * H_ + 255) / 256, 256, 0, stream>>>(z, Wzp, bzp, zpF);
  gbias0_kernel<<<(B_ * G4_ + 255) / 256, 256, 0, stream>>>(zpF, Wih0, bih0, bhh0, gb0T);
  cbias1_kernel<<<(G4_ + 255) / 256, 256, 0, stream>>>(bih1, bhh1, cb1);

  decoder_kernel<<<B_ / 16, 512, 0, stream>>>(
      Wih0h, Whh0h, Wih1h, Whh1h, Wpreh, Wouth,
      gb0T, cb1, bpre, bout, lng, lnb,
      haF, caF, hbF, cbF, (float*)d_out);
}
